// Contrastive_FeatureExtractor_conv_48258252537957
// MI455X (gfx1250) — matrix operand order repaired and checked by reading, not compiled, not run on hardware
//
#include <hip/hip_runtime.h>
#include <hip/hip_bf16.h>

// ---------------------------------------------------------------------------
// MI455X (gfx1250) implementation.
// Pipeline (all on one stream, fixed order, fully deterministic, no atomics):
//   1. conv1(+relu) -> h1 stored channel-last bf16  [N][8052][32]
//   2. stats(h1) -> BN1 folded into conv2 weights (B-fragment-swizzled bf16)
//   3. conv2 as WMMA GEMM (M=687104,K=384) + bias + relu -> h2 bf16 [N][671][32]
//   4. stats(h2) -> BN2 folded into conv3 weights
//   5. conv3 as WMMA GEMM (M=27648,K=768) -> h3 f32 [N][27][32]
//   6. gather 13 cols/sample -> mu/sd/max -> f_raw [N][96]
//   7. stats(f_raw) -> BN3 folded into fc1; fc1+relu (VALU, tiny) -> f [N][32]
//   8. stats(f) -> BN4 affine applied -> f_n f32 + B-swizzled bf16
//   9. A -> A^T bf16 (tiled transpose); x1 = A^T @ f_n via WMMA GEMM
//  10. out = [f_n | x1] @ gcn_w^T + gcn_b (VALU, tiny)
// ---------------------------------------------------------------------------

typedef __attribute__((ext_vector_type(16))) __bf16 v16bf;
typedef __attribute__((ext_vector_type(8)))  float  v8f;

union ABPack { uint4 q[2]; v16bf v; };   // two 16B global loads -> one A fragment

__device__ __forceinline__ unsigned short bf16_rne(float f) {
    unsigned u = __float_as_uint(f);
    u += 0x7FFFu + ((u >> 16) & 1u);     // round-to-nearest-even
    return (unsigned short)(u >> 16);
}
__device__ __forceinline__ float bf2f(unsigned short u) {
    return __uint_as_float(((unsigned)u) << 16);
}

// ---------------- conv1: 1->32 ch, K=13, stride 1, +relu, store bf16 NLC ----
__global__ void k_conv1(const float* __restrict__ x, const float* __restrict__ w1,
                        const float* __restrict__ b1, unsigned short* __restrict__ h1) {
    __shared__ float sw[32 * 13];
    __shared__ float sb[32];
    const int n = blockIdx.x;
    for (int i = threadIdx.x; i < 32 * 13; i += blockDim.x) sw[i] = w1[i];
    if (threadIdx.x < 32) sb[threadIdx.x] = b1[threadIdx.x];
    __syncthreads();
    const float* xr = x + (size_t)n * 8064;
    unsigned short* hr = h1 + (size_t)n * 8052 * 32;
    for (int q = threadIdx.x; q < 8052; q += blockDim.x) {
        float xv[13];
#pragma unroll
        for (int j = 0; j < 13; ++j) xv[j] = xr[q + j];
#pragma unroll
        for (int c = 0; c < 32; ++c) {
            float a = sb[c];
#pragma unroll
            for (int j = 0; j < 13; ++j) a = fmaf(xv[j], sw[c * 13 + j], a);
            hr[(size_t)q * 32 + c] = bf16_rne(fmaxf(a, 0.f));
        }
    }
}

// -------- per-sample sum/sumsq over bf16 [N][P][32] -> part[N][32][2] -------
__global__ void k_stats(const unsigned short* __restrict__ h, int P,
                        float* __restrict__ part) {
    __shared__ float ls[8][32][2];
    const int n = blockIdx.x;
    const int c = threadIdx.x & 31, g = threadIdx.x >> 5;
    const unsigned short* hr = h + (size_t)n * P * 32;
    float s = 0.f, ss = 0.f;
    for (int p = g; p < P; p += 8) {
        float v = bf2f(hr[(size_t)p * 32 + c]);
        s += v; ss += v * v;
    }
    ls[g][c][0] = s; ls[g][c][1] = ss;
    __syncthreads();
    if (threadIdx.x < 32) {
        float a = 0.f, b = 0.f;
        for (int g2 = 0; g2 < 8; ++g2) { a += ls[g2][threadIdx.x][0]; b += ls[g2][threadIdx.x][1]; }
        part[((size_t)n * 32 + threadIdx.x) * 2 + 0] = a;
        part[((size_t)n * 32 + threadIdx.x) * 2 + 1] = b;
    }
}

// ---- finalize BN(32ch) from partials; fold into conv weights (B-swizzle) ---
// W layout produced: [KT tiles][2 col-halves][32 lanes][16 bf16] matching the
// B fragment in this code's contiguous k order (same as A, not AMD's table order): lane L -> col (L&15)+16h, K = 32t+16*(L>>4)+j
__global__ void k_prep_conv(const float* __restrict__ part, float invcnt,
                            const float* __restrict__ gamma, const float* __restrict__ beta,
                            const float* __restrict__ w, const float* __restrict__ bsrc,
                            int Q, unsigned short* __restrict__ wsw, float* __restrict__ bfold) {
    __shared__ float s_sc[32], s_sh[32];
    const int tid = threadIdx.x;
    if (tid < 32) {
        float s = 0.f, ss = 0.f;
        for (int n = 0; n < 1024; ++n) {
            s  += part[((size_t)n * 32 + tid) * 2 + 0];
            ss += part[((size_t)n * 32 + tid) * 2 + 1];
        }
        float m = s * invcnt;
        float v = ss * invcnt - m * m;
        float sc = gamma[tid] / sqrtf(v + 1e-5f);
        s_sc[tid] = sc;
        s_sh[tid] = beta[tid] - m * sc;
    }
    __syncthreads();
    const int tot = Q * 2 * 32 * 16;               // K = 32*Q, KT = Q tiles
    for (int idx = tid; idx < tot; idx += blockDim.x) {
        int j = idx & 15, L = (idx >> 4) & 31, h = (idx >> 9) & 1, t = idx >> 10;
        int k  = 32 * t + ((L >> 4) << 4) + j;     // K index
        int q  = k >> 5, ic = k & 31;              // K = q*32 + ic (channel-last rows)
        int oc = (h << 4) + (L & 15);
        wsw[idx] = bf16_rne(w[(size_t)(oc * 32 + ic) * Q + q] * s_sc[ic]);
    }
    if (tid < 32) {
        int oc = tid;
        float acc = bsrc[oc];
        for (int ic = 0; ic < 32; ++ic) {
            float sm = 0.f;
            for (int q = 0; q < Q; ++q) sm += w[(size_t)(oc * 32 + ic) * Q + q];
            acc += s_sh[ic] * sm;
        }
        bfold[oc] = acc;
    }
}

// ------------------- generic WMMA GEMM: [Mrows x K] * [K x 32] -------------
// A: bf16 rows; row r lives at  n*slab + p*K  with n = r/rowsPerN, p = r%rowsPerN.
// B: pre-swizzled [KT][2][32][16] bf16.  mode 1: +bias,relu,store bf16;
// mode 0: +bias(optional), store f32.  8 waves/block, 1 M-tile (16 rows)/wave.
__global__ void __launch_bounds__(256)
k_gemm(const unsigned short* __restrict__ Abf, const unsigned short* __restrict__ Bsw,
       const float* __restrict__ bias, int Mrows, int K, int KT,
       int rowsPerN, long long slab, int mode,
       unsigned short* __restrict__ outb, float* __restrict__ outf) {
    const int wave = threadIdx.x >> 5;
    const int lane = threadIdx.x & 31;
    const int mt = blockIdx.x * 8 + wave;
    if (mt * 16 >= Mrows) return;                 // wave-uniform: EXEC stays full
    const int row0 = mt * 16;
    const int r  = lane & 15;
    const int hi = lane >> 4;
    const int row = row0 + r;
    const int nn = row / rowsPerN;
    const int pp = row - nn * rowsPerN;
    const unsigned short* arow = Abf + (size_t)nn * (size_t)slab + (size_t)pp * K + hi * 16;

    v8f acc0 = {0.f, 0.f, 0.f, 0.f, 0.f, 0.f, 0.f, 0.f};
    v8f acc1 = acc0;
    for (int t = 0; t < KT; ++t) {
        ABPack ap;
        ap.q[0] = *(const uint4*)(arow + 32 * t);        // K = 32t + 16*hi   .. +7
        ap.q[1] = *(const uint4*)(arow + 32 * t + 8);    // K = 32t+16*hi+8 .. +15
        v16bf b0 = *(const v16bf*)(Bsw + ((size_t)(t * 2 + 0) * 32 + lane) * 16);
        v16bf b1 = *(const v16bf*)(Bsw + ((size_t)(t * 2 + 1) * 32 + lane) * 16);
        acc0 = __builtin_amdgcn_wmma_f32_16x16x32_bf16(false, ap.v, false, b0,
                                                       (short)0, acc0, false, false);
        acc1 = __builtin_amdgcn_wmma_f32_16x16x32_bf16(false, ap.v, false, b1,
                                                       (short)0, acc1, false, false);
    }
    const int col = lane & 15;
    const float bv0 = bias ? bias[col]      : 0.f;
    const float bv1 = bias ? bias[16 + col] : 0.f;
#pragma unroll
    for (int e = 0; e < 8; ++e) {
        int m = row0 + e + 8 * hi;                // D layout: VGPR e -> row e+8*(lane>=16)
        float v0 = acc0[e] + bv0;
        float v1 = acc1[e] + bv1;
        if (mode == 1) {
            outb[(size_t)m * 32 + col]      = bf16_rne(fmaxf(v0, 0.f));
            outb[(size_t)m * 32 + 16 + col] = bf16_rne(fmaxf(v1, 0.f));
        } else {
            outf[(size_t)m * 32 + col]      = v0;
            outf[(size_t)m * 32 + 16 + col] = v1;
        }
    }
}

// ------- per-sample gather of 13 columns -> mu / unbiased sd / max ----------
__global__ void k_gather(const float* __restrict__ h3, const int* __restrict__ perm,
                         float* __restrict__ fraw) {
    const int n = blockIdx.x, c = threadIdx.x;
    const float* hb = h3 + (size_t)n * 27 * 32;
    const int* pi = perm + n * 13;
    float s = 0.f, ss = 0.f, mx = -3.402823466e38f;
#pragma unroll
    for (int j = 0; j < 13; ++j) {
        float v = hb[pi[j] * 32 + c];
        s += v; ss += v * v; mx = fmaxf(mx, v);
    }
    float mu  = s * (1.f / 13.f);
    float var = (ss - 13.f * mu * mu) * (1.f / 12.f);   // ddof=1
    fraw[n * 96 + c]      = mu;
    fraw[n * 96 + 32 + c] = sqrtf(fmaxf(var, 0.f));
    fraw[n * 96 + 64 + c] = mx;
}

// ------- BN3 stats over [1024][96]; fold affine into fc1 weights/bias -------
__global__ void k_prep_fc1(const float* __restrict__ fraw, const float* __restrict__ g,
                           const float* __restrict__ b, const float* __restrict__ fc1w,
                           const float* __restrict__ fc1b,
                           float* __restrict__ wp, float* __restrict__ bp) {
    __shared__ float sc[96], sh[96];
    const int tid = threadIdx.x;
    if (tid < 96) {
        float s = 0.f, ss = 0.f;
        for (int n = 0; n < 1024; ++n) { float v = fraw[n * 96 + tid]; s += v; ss += v * v; }
        float m = s * (1.f / 1024.f), var = ss * (1.f / 1024.f) - m * m;
        float scale = g[tid] / sqrtf(var + 1e-5f);
        sc[tid] = scale; sh[tid] = b[tid] - m * scale;
    }
    __syncthreads();
    for (int idx = tid; idx < 32 * 96; idx += blockDim.x) {
        int j = idx % 96;
        wp[idx] = fc1w[idx] * sc[j];
    }
    if (tid < 32) {
        float a = fc1b[tid];
        for (int j = 0; j < 96; ++j) a = fmaf(fc1w[tid * 96 + j], sh[j], a);
        bp[tid] = a;
    }
}

// ------------------------- fc1 (96->32) + relu, VALU ------------------------
__global__ void k_fc1(const float* __restrict__ fraw, const float* __restrict__ wp,
                      const float* __restrict__ bp, float* __restrict__ f) {
    int idx = blockIdx.x * blockDim.x + threadIdx.x;
    if (idx >= 1024 * 32) return;
    int n = idx >> 5, o = idx & 31;
    const float* fr = fraw + n * 96;
    const float* wr = wp + o * 96;
    float a = bp[o];
#pragma unroll 8
    for (int j = 0; j < 96; ++j) a = fmaf(fr[j], wr[j], a);
    f[idx] = fmaxf(a, 0.f);
}

// --------------------- BN4 stats (32 ch over N=1024) ------------------------
__global__ void k_prep_gcn(const float* __restrict__ f, const float* __restrict__ g,
                           const float* __restrict__ b, float* __restrict__ s4,
                           float* __restrict__ t4) {
    const int c = threadIdx.x;
    if (c >= 32) return;
    float s = 0.f, ss = 0.f;
    for (int n = 0; n < 1024; ++n) { float v = f[n * 32 + c]; s += v; ss += v * v; }
    float m = s * (1.f / 1024.f), var = ss * (1.f / 1024.f) - m * m;
    float sc = g[c] / sqrtf(var + 1e-5f);
    s4[c] = sc; t4[c] = b[c] - m * sc;
}

// -- apply BN4: f_n f32 [1024][32] and B-fragment-swizzled bf16 for the GEMM -
__global__ void k_bn4(const float* __restrict__ f, const float* __restrict__ s4,
                      const float* __restrict__ t4, float* __restrict__ fn,
                      unsigned short* __restrict__ fsw) {
    int idx = blockIdx.x * blockDim.x + threadIdx.x;
    if (idx >= 32768) return;
    { int c = idx & 31; fn[idx] = f[idx] * s4[c] + t4[c]; }
    {   // fsw[t][h][L][j]:  K-row = 32t+16*(L>>4)+j, col = 16h+(L&15)
        int j = idx & 15, L = (idx >> 4) & 31, h = (idx >> 9) & 1, t = idx >> 10;
        int row = 32 * t + ((L >> 4) << 4) + j;
        int c = (h << 4) + (L & 15);
        fsw[idx] = bf16_rne(f[row * 32 + c] * s4[c] + t4[c]);
    }
}

// --------------- A (f32, [v][w]) -> A^T (bf16, [w][v]) tiled ----------------
__global__ void k_transA(const float* __restrict__ A, unsigned short* __restrict__ At) {
    __shared__ unsigned short tile[32][33];
    const int bx = blockIdx.x * 32, by = blockIdx.y * 32;
    for (int yy = threadIdx.y; yy < 32; yy += 8)
        tile[yy][threadIdx.x] = bf16_rne(A[(size_t)(by + yy) * 1024 + bx + threadIdx.x]);
    __syncthreads();
    for (int yy = threadIdx.y; yy < 32; yy += 8)
        At[(size_t)(bx + yy) * 1024 + by + threadIdx.x] = tile[threadIdx.x][yy];
}

// --------- out = f_n @ Wf^T + x1 @ Wx^T + b   (gcn_w = [Wf | Wx]) ----------
__global__ void k_final(const float* __restrict__ fn, const float* __restrict__ x1,
                        const float* __restrict__ gw, const float* __restrict__ gb,
                        float* __restrict__ out) {
    int idx = blockIdx.x * blockDim.x + threadIdx.x;
    if (idx >= 32768) return;
    int w = idx >> 5, o = idx & 31;
    const float* fr = fn + w * 32;
    const float* xr = x1 + w * 32;
    const float* w0 = gw + o * 64;
    float a = gb[o];
#pragma unroll
    for (int c = 0; c < 32; ++c) a = fmaf(fr[c], w0[c], a);
#pragma unroll
    for (int c = 0; c < 32; ++c) a = fmaf(xr[c], w0[32 + c], a);
    out[idx] = a;
}

// ---------------------------------------------------------------------------
extern "C" void kernel_launch(void* const* d_in, const int* in_sizes, int n_in,
                              void* d_out, int out_size, void* d_ws, size_t ws_size,
                              hipStream_t stream) {
    const float* x       = (const float*)d_in[0];
    const int*   perm    = (const int*)  d_in[1];
    const float* A       = (const float*)d_in[2];
    const float* conv1_w = (const float*)d_in[3];
    const float* conv1_b = (const float*)d_in[4];
    const float* conv2_w = (const float*)d_in[5];
    const float* conv2_b = (const float*)d_in[6];
    const float* conv3_w = (const float*)d_in[7];
    const float* conv3_b = (const float*)d_in[8];
    const float* bn1_g = (const float*)d_in[9],  *bn1_b = (const float*)d_in[10];
    const float* bn2_g = (const float*)d_in[11], *bn2_b = (const float*)d_in[12];
    const float* bn3_g = (const float*)d_in[13], *bn3_b = (const float*)d_in[14];
    const float* bn4_g = (const float*)d_in[15], *bn4_b = (const float*)d_in[16];
    const float* fc1_w = (const float*)d_in[17], *fc1_b = (const float*)d_in[18];
    const float* gcn_w = (const float*)d_in[19], *gcn_b = (const float*)d_in[20];
    float* out = (float*)d_out;

    // -------- workspace carve-up (256B aligned) --------
    char* base = (char*)d_ws;
    size_t off = 0;
    auto carve = [&](size_t bytes) -> char* {
        char* p = base + off;
        off = (off + bytes + 255) & ~(size_t)255;
        return p;
    };
    unsigned short* H1   = (unsigned short*)carve((size_t)1024 * 8052 * 32 * 2); // 503 MiB
    unsigned short* H2   = (unsigned short*)carve((size_t)1024 * 671 * 32 * 2);  //  42 MiB
    float*          H3   = (float*)         carve((size_t)1024 * 27 * 32 * 4);
    float*          PART = (float*)         carve((size_t)1024 * 32 * 2 * 4);    // reused
    unsigned short* W2SW = (unsigned short*)carve((size_t)12 * 2 * 32 * 16 * 2);
    float*          B2P  = (float*)         carve(32 * 4);
    unsigned short* W3SW = (unsigned short*)carve((size_t)24 * 2 * 32 * 16 * 2);
    float*          B3P  = (float*)         carve(32 * 4);
    float*          FRAW = (float*)         carve((size_t)1024 * 96 * 4);
    float*          FC1P = (float*)         carve((size_t)32 * 96 * 4);
    float*          BFCP = (float*)         carve(32 * 4);
    float*          F    = (float*)         carve((size_t)1024 * 32 * 4);
    float*          S4   = (float*)         carve(32 * 4);
    float*          T4   = (float*)         carve(32 * 4);
    float*          FN   = (float*)         carve((size_t)1024 * 32 * 4);
    unsigned short* FSW  = (unsigned short*)carve((size_t)32 * 2 * 32 * 16 * 2);
    unsigned short* AT   = (unsigned short*)carve((size_t)1024 * 1024 * 2);
    float*          X1   = (float*)         carve((size_t)1024 * 32 * 4);
    (void)ws_size; (void)n_in; (void)in_sizes; (void)out_size;

    // 1) conv1 + relu -> bf16 channel-last
    k_conv1<<<1024, 256, 0, stream>>>(x, conv1_w, conv1_b, H1);
    // 2) BN1 stats; fold into conv2 weights
    k_stats<<<1024, 256, 0, stream>>>(H1, 8052, PART);
    k_prep_conv<<<1, 256, 0, stream>>>(PART, 1.0f / (1024.0f * 8052.0f),
                                       bn1_g, bn1_b, conv2_w, conv2_b, 12, W2SW, B2P);
    // 3) conv2 as WMMA GEMM: M=687104, K=384, relu + bf16 out
    k_gemm<<<5368, 256, 0, stream>>>(H1, W2SW, B2P, 687104, 384, 12,
                                     671, (long long)8052 * 32, 1, H2, nullptr);
    // 4) BN2 stats; fold into conv3 weights
    k_stats<<<1024, 256, 0, stream>>>(H2, 671, PART);
    k_prep_conv<<<1, 256, 0, stream>>>(PART, 1.0f / (1024.0f * 671.0f),
                                       bn2_g, bn2_b, conv3_w, conv3_b, 24, W3SW, B3P);
    // 5) conv3 as WMMA GEMM: M=27648, K=768, f32 out
    k_gemm<<<216, 256, 0, stream>>>(H2, W3SW, B3P, 27648, 768, 24,
                                    27, (long long)671 * 32, 0, nullptr, H3);
    // 6) gather + mu/sd/max -> f_raw [1024][96]
    k_gather<<<1024, 32, 0, stream>>>(H3, perm, FRAW);
    // 7) BN3 fold into fc1; fc1 + relu
    k_prep_fc1<<<1, 256, 0, stream>>>(FRAW, bn3_g, bn3_b, fc1_w, fc1_b, FC1P, BFCP);
    k_fc1<<<128, 256, 0, stream>>>(FRAW, FC1P, BFCP, F);
    // 8) BN4 stats + apply (f32 + B-swizzled bf16)
    k_prep_gcn<<<1, 32, 0, stream>>>(F, bn4_g, bn4_b, S4, T4);
    k_bn4<<<128, 256, 0, stream>>>(F, S4, T4, FN, FSW);
    // 9) A^T in bf16; x1 = A^T @ f_n via WMMA GEMM (M=1024, K=1024)
    k_transA<<<dim3(32, 32), dim3(32, 8), 0, stream>>>(A, AT);
    k_gemm<<<8, 256, 0, stream>>>(AT, FSW, nullptr, 1024, 1024, 32,
                                  1024, 0LL, 0, nullptr, X1);
    // 10) final projection -> d_out [1024][32]
    k_final<<<128, 256, 0, stream>>>(FN, X1, gcn_w, gcn_b, out);
}
